// GCN_41437844472432
// MI455X (gfx1250) — matrix operand order repaired and checked by reading, not compiled, not run on hardware
//
#include <hip/hip_runtime.h>
#include <hip/hip_bf16.h>

// ---------------------------------------------------------------------------
// GCN forward for MI455X (gfx1250, wave32):
//   h1 = relu(GCNConv(x, W1, b1));  h2 = relu(GCNConv(h1, W2, b2))
//   out = mean_pool(h2, batch) @ Wh + bh
// Edge aggregation is bandwidth/atomic bound (~76.8M f32 atomics per layer);
// the [100k,64]x[64,64] GEMM uses v_wmma_f32_16x16x32_bf16 with W2 staged
// into LDS via global_load_async_to_lds_b128 (ASYNCcnt path).
// ---------------------------------------------------------------------------

typedef __attribute__((ext_vector_type(16))) __bf16 v16bf;
typedef __attribute__((ext_vector_type(8)))  float  v8f;

#define HID 64

// ---------------- degree / normalization ----------------
__global__ void deg_kernel(const int* __restrict__ dst, float* __restrict__ deg, int E) {
    int e = blockIdx.x * blockDim.x + threadIdx.x;
    if (e < E) atomicAdd(&deg[dst[e]], 1.0f);
}

__global__ void dinv_kernel(float* __restrict__ deg, int N) {
    int i = blockIdx.x * blockDim.x + threadIdx.x;
    if (i < N) deg[i] = rsqrtf(deg[i] + 1.0f);   // self-loop adds 1
}

// ---------------- layer 1 GEMM: [N,7] @ [7,64] (K too small for WMMA) -------
__global__ void gemm_xw1(const float* __restrict__ x, const float* __restrict__ W1,
                         float* __restrict__ hlin, int N) {
    int t = blockIdx.x * blockDim.x + threadIdx.x;
    int i = t >> 6, c = t & 63;
    if (i >= N) return;
    const float* xr = x + (size_t)i * 7;
    float acc = 0.f;
#pragma unroll
    for (int k = 0; k < 7; ++k) acc = fmaf(xr[k], W1[k * HID + c], acc);
    hlin[(size_t)i * HID + c] = acc;
}

// ---------------- edge scatter-gather aggregation ----------------
// 16 threads per edge; each handles 4 channels via a b128 gather.
__global__ void edge_agg(const int* __restrict__ src, const int* __restrict__ dst,
                         const float* __restrict__ dinv, const float* __restrict__ h,
                         float* __restrict__ agg, int E) {
    long long t = (long long)blockIdx.x * blockDim.x + threadIdx.x;
    int e  = (int)(t >> 4);
    int c4 = (int)(t & 15) * 4;
    if (e >= E) return;
    int s = src[e], d = dst[e];
    float coef = dinv[s] * dinv[d];
    const float4* hv = (const float4*)(h + (size_t)s * HID + c4);
    __builtin_prefetch(hv, 0, 1);                    // global_prefetch_b8
    float4 v = *hv;
    float* arow = agg + (size_t)d * HID + c4;
    atomicAdd(arow + 0, coef * v.x);
    atomicAdd(arow + 1, coef * v.y);
    atomicAdd(arow + 2, coef * v.z);
    atomicAdd(arow + 3, coef * v.w);
}

// ---------------- self-loop + bias + relu (out may alias hlin) --------------
__global__ void finish_layer(const float* __restrict__ agg, const float* __restrict__ hlin,
                             const float* __restrict__ dinv, const float* __restrict__ b,
                             float* __restrict__ out, int N) {
    int t = blockIdx.x * blockDim.x + threadIdx.x;
    int i = t >> 6, c = t & 63;
    if (i >= N) return;
    size_t idx = (size_t)i * HID + c;
    float di = dinv[i];
    float v = agg[idx] + di * di * hlin[idx] + b[c];
    out[idx] = fmaxf(v, 0.0f);
}

// ---------------- layer 2 GEMM: [N,64] @ [64,64] via bf16 WMMA --------------
// Block = 256 threads = 8 waves. Wave w computes output tile
//   (mtile = blockIdx.x*2 + w/4, ntile = w%4), i.e. a 32x64 slab per block.
// W2 (fp32, 16 KB) staged into LDS with global_load_async_to_lds_b128;
// converted to bf16 at fragment-build time.
__global__ void gemm_wmma_bf16(const float* __restrict__ h, const float* __restrict__ W2f,
                               float* __restrict__ out, int nrows) {
    __shared__ float Wlds[HID * HID];                // 16 KB

    // Async-stage W2 into LDS: 4 passes x 256 threads x 16B = 16 KB.
    unsigned lds_base = (unsigned)(unsigned long long)(&Wlds[0]);
#pragma unroll
    for (int p = 0; p < 4; ++p) {
        unsigned     loff = lds_base + (unsigned)((p * 256 + threadIdx.x) * 16);
        const float* gptr = W2f + (p * 256 + threadIdx.x) * 4;
        asm volatile("global_load_async_to_lds_b128 %0, %1, off"
                     :: "v"(loff), "v"(gptr) : "memory");
    }
    asm volatile("s_wait_asynccnt 0x0" ::: "memory");
    __syncthreads();

    int wave = threadIdx.x >> 5;
    int lane = threadIdx.x & 31;
    int mtile = blockIdx.x * 2 + (wave >> 2);
    int ntile = wave & 3;
    if (mtile * 16 >= nrows) return;                 // wave-uniform guard

    int half = lane >> 4;                            // 0: lanes 0-15, 1: lanes 16-31
    int m = lane & 15;                               // A row / B-C column index
    int row = mtile * 16 + m;
    if (row >= nrows) row = nrows - 1;               // safety clamp (N%16==0 normally)

    v8f acc = {};
#pragma unroll
    for (int kb = 0; kb < HID; kb += 32) {
        // A tile 16x32 bf16: lane holds M=m, K = kb + half*8 + {0..7}
        //                                and K = kb + 16 + half*8 + {0..7}
        v16bf a;
        const float* hrow = h + (size_t)row * HID + kb + half * 8;
#pragma unroll
        for (int j = 0; j < 8; ++j) a[j] = (__bf16)hrow[j];
#pragma unroll
        for (int j = 0; j < 8; ++j) a[8 + j] = (__bf16)hrow[16 + j];

        // B tile 32x16 bf16: element s holds K = kb + half*8 + s (s<8),
        // K = kb + 16 + half*8 + (s-8) (s>=8), N = lane%16.
        v16bf bm;
        const float* wcol = Wlds + (size_t)(kb + half * 8) * HID + ntile * 16 + m;
#pragma unroll
        for (int v = 0; v < 8; ++v) {
            bm[2 * v]     = (__bf16)wcol[(2 * v + (v >> 2) * 8) * HID];
            bm[2 * v + 1] = (__bf16)wcol[(2 * v + 1 + (v >> 2) * 8) * HID];
        }

        acc = __builtin_amdgcn_wmma_f32_16x16x32_bf16(
            /*neg_a=*/false, a, /*neg_b=*/false, bm,
            /*c_mod=*/(short)0, acc, /*reuse_a=*/false, /*reuse_b=*/false);
    }

    // C/D layout: VGPR r -> M = r + 8*half, N = lane%16
    float* orow = out + (size_t)(mtile * 16 + half * 8) * HID + ntile * 16 + m;
#pragma unroll
    for (int r = 0; r < 8; ++r) orow[(size_t)r * HID] = acc[r];
}

// ---------------- mean pool (segment sums); 16 threads/node, float4 --------
__global__ void pool_kernel(const float* __restrict__ h, const int* __restrict__ batch,
                            float* __restrict__ pool, float* __restrict__ cnt, int N) {
    long long t = (long long)blockIdx.x * blockDim.x + threadIdx.x;
    int i  = (int)(t >> 4);
    int c4 = (int)(t & 15) * 4;
    if (i >= N) return;
    int g = batch[i];
    float4 v = *(const float4*)(h + (size_t)i * HID + c4);
    float* prow = pool + (size_t)g * HID + c4;
    atomicAdd(prow + 0, v.x);
    atomicAdd(prow + 1, v.y);
    atomicAdd(prow + 2, v.z);
    atomicAdd(prow + 3, v.w);
    if (c4 == 0) atomicAdd(&cnt[g], 1.0f);
}

// ---------------- classifier head: [G,64] @ [64,2] ----------------
__global__ void head_kernel(const float* __restrict__ pool, const float* __restrict__ cnt,
                            const float* __restrict__ Wh, const float* __restrict__ bh,
                            float* __restrict__ out, int G) {
    int t = blockIdx.x * blockDim.x + threadIdx.x;
    int g = t >> 1, j = t & 1;
    if (g >= G) return;
    float inv = 1.0f / fmaxf(cnt[g], 1.0f);
    const float* pr = pool + (size_t)g * HID;
    float acc = 0.f;
#pragma unroll
    for (int k = 0; k < HID; ++k) acc = fmaf(pr[k], Wh[k * 2 + j], acc);
    out[(size_t)g * 2 + j] = acc * inv + bh[j];
}

// ---------------------------------------------------------------------------
extern "C" void kernel_launch(void* const* d_in, const int* in_sizes, int n_in,
                              void* d_out, int out_size, void* d_ws, size_t ws_size,
                              hipStream_t stream) {
    const float* x  = (const float*)d_in[0];
    const int*   ei = (const int*)  d_in[1];   // [2, E]
    const int*   bt = (const int*)  d_in[2];   // [N]
    const float* W1 = (const float*)d_in[3];
    const float* b1 = (const float*)d_in[4];
    const float* W2 = (const float*)d_in[5];
    const float* b2 = (const float*)d_in[6];
    const float* Wh = (const float*)d_in[7];
    const float* bh = (const float*)d_in[8];
    float* out = (float*)d_out;

    const int E = in_sizes[1] / 2;
    const int N = in_sizes[2];
    const int G = out_size / 2;
    const int* src = ei;
    const int* dst = ei + E;

    // workspace carve-up (floats)
    float* ws = (float*)d_ws;
    size_t nAlign = ((size_t)N + 511) & ~(size_t)511;
    size_t NH = (size_t)N * HID;
    float* dinv = ws;                        // N      (starts as degree accumulator)
    float* bufA = dinv + nAlign;             // N*64   h1lin -> h1
    float* bufB = bufA + NH;                 // N*64   agg1 -> h2lin -> h2
    float* bufC = bufB + NH;                 // N*64   agg2
    float* pool = bufC + NH;                 // G*64
    float* cnt  = pool + (size_t)G * HID;    // G

    // zero the accumulators (graph-capturable memset nodes)
    hipMemsetAsync(dinv, 0, (size_t)N * sizeof(float), stream);
    hipMemsetAsync(bufB, 0, NH * sizeof(float), stream);
    hipMemsetAsync(bufC, 0, NH * sizeof(float), stream);
    hipMemsetAsync(pool, 0, ((size_t)G * HID + G) * sizeof(float), stream);

    const int T = 256;
    // degree + D^{-1/2}
    deg_kernel <<<(E + T - 1) / T, T, 0, stream>>>(dst, dinv, E);
    dinv_kernel<<<(N + T - 1) / T, T, 0, stream>>>(dinv, N);

    // layer 1
    long long nh   = (long long)N * HID;
    long long eh16 = (long long)E * 16;
    gemm_xw1 <<<(int)((nh + T - 1) / T), T, 0, stream>>>(x, W1, bufA, N);
    edge_agg <<<(int)((eh16 + T - 1) / T), T, 0, stream>>>(src, dst, dinv, bufA, bufB, E);
    finish_layer<<<(int)((nh + T - 1) / T), T, 0, stream>>>(bufB, bufA, dinv, b1, bufA, N); // h1 -> bufA

    // layer 2 (WMMA GEMM), agg into bufC
    gemm_wmma_bf16<<<(N + 31) / 32, 256, 0, stream>>>(bufA, W2, bufB, N);                    // h2lin -> bufB
    edge_agg <<<(int)((eh16 + T - 1) / T), T, 0, stream>>>(src, dst, dinv, bufB, bufC, E);
    finish_layer<<<(int)((nh + T - 1) / T), T, 0, stream>>>(bufC, bufB, dinv, b2, bufB, N); // h2 -> bufB

    // pool + head
    long long nh16 = (long long)N * 16;
    pool_kernel<<<(int)((nh16 + T - 1) / T), T, 0, stream>>>(bufB, bt, pool, cnt, N);
    head_kernel<<<(G * 2 + T - 1) / T, T, 0, stream>>>(pool, cnt, Wh, bh, out, G);
}
